// S6_42528766165488
// MI455X (gfx1250) — matrix operand order repaired and checked by reading, not compiled, not run on hardware
//
#include <hip/hip_runtime.h>
#include <math.h>

// ---- problem constants (match reference) ----
#define D_MODEL  1024
#define D_STATE  16
#define D_CONV   4
#define D_INNER  2048          // EXPAND * D_MODEL
#define DT_RANK  128           // ceil(D_INNER/16)
#define NBATCH   4
#define LSEQ     2048
#define NROWS    (NBATCH * LSEQ)          // 8192 token rows
#define XPROJ_N  (DT_RANK + 2 * D_STATE)  // 160 (real)
#define XPROJ_LD 192                      // padded to a multiple of 64
#define NCHUNK   32                       // scan chunks per sequence
#define CLEN     (LSEQ / NCHUNK)          // 64 steps per chunk

typedef __attribute__((ext_vector_type(16))) _Float16 v16h;
typedef __attribute__((ext_vector_type(8)))  float    v8f;

__device__ __forceinline__ float sigmoid_f(float x) { return 1.0f / (1.0f + __expf(-x)); }
__device__ __forceinline__ float softplus_f(float x) {
  return (x > 20.f) ? x : log1pf(__expf(x));
}

// ============================================================
// RMSNorm: one 256-thread block per row of D_MODEL=1024 floats
// ============================================================
__global__ void rmsnorm_kernel(const float* __restrict__ x,
                               const float* __restrict__ w,
                               float* __restrict__ out) {
  const int row = blockIdx.x;
  const float* xr = x + (size_t)row * D_MODEL;
  float v[4];
  float ss = 0.f;
#pragma unroll
  for (int i = 0; i < 4; i++) {
    v[i] = xr[threadIdx.x + 256 * i];
    ss += v[i] * v[i];
  }
  __shared__ float red[256];
  red[threadIdx.x] = ss;
  __syncthreads();
  for (int s = 128; s > 0; s >>= 1) {
    if (threadIdx.x < s) red[threadIdx.x] += red[threadIdx.x + s];
    __syncthreads();
  }
  const float scale = rsqrtf(red[0] * (1.0f / (float)D_MODEL) + 1e-5f);
  float* orow = out + (size_t)row * D_MODEL;
#pragma unroll
  for (int i = 0; i < 4; i++) {
    const int c = threadIdx.x + 256 * i;
    orow[c] = v[i] * scale * w[c];
  }
}

// ============================================================
// Residual add + RMSNorm (final)
// ============================================================
__global__ void add_rmsnorm_kernel(const float* __restrict__ c,
                                   const float* __restrict__ resid,
                                   const float* __restrict__ w,
                                   float* __restrict__ out) {
  const int row = blockIdx.x;
  const float* cr = c + (size_t)row * D_MODEL;
  const float* rr = resid + (size_t)row * D_MODEL;
  float v[4];
  float ss = 0.f;
#pragma unroll
  for (int i = 0; i < 4; i++) {
    const int col = threadIdx.x + 256 * i;
    v[i] = cr[col] + rr[col];
    ss += v[i] * v[i];
  }
  __shared__ float red[256];
  red[threadIdx.x] = ss;
  __syncthreads();
  for (int s = 128; s > 0; s >>= 1) {
    if (threadIdx.x < s) red[threadIdx.x] += red[threadIdx.x + s];
    __syncthreads();
  }
  const float scale = rsqrtf(red[0] * (1.0f / (float)D_MODEL) + 1e-5f);
  float* orow = out + (size_t)row * D_MODEL;
#pragma unroll
  for (int i = 0; i < 4; i++) {
    const int col = threadIdx.x + 256 * i;
    orow[col] = v[i] * scale * w[col];
  }
}

// ============================================================
// Zero-pad x_proj_w [160,2048] -> [192,2048] so the GEMM tiles exactly.
// ============================================================
__global__ void pad_w_kernel(const float* __restrict__ src, float* __restrict__ dst) {
  const int idx = blockIdx.x * 256 + threadIdx.x;   // over 192*2048
  if (idx >= XPROJ_LD * D_INNER) return;
  const int row = idx / D_INNER;
  dst[idx] = (row < XPROJ_N) ? src[idx] : 0.f;
}

// ============================================================
// WMMA GEMM:  C[M,N] = A[M,K] @ W[N,K]^T   (f32 in, f16 WMMA, f32 acc)
// One wave -> 32(M) x 64(N) tile: 2 A frags x 4 B frags -> 8 accumulators.
// No predication (M%32==0, N%64==0, K%32==0).
// Fragment k order: contiguous, lane half h holds k = 16h..16h+15 in both A and B (wave32).
// ============================================================
__global__ void gemm_wmma_kernel(const float* __restrict__ A,
                                 const float* __restrict__ W,
                                 float* __restrict__ C,
                                 int K, int lda, int ldw, int ldc,
                                 int total_waves, int tiles_n) {
  const int wave = blockIdx.x * (blockDim.x >> 5) + (threadIdx.x >> 5);
  if (wave >= total_waves) return;  // wave-uniform exit, before any WMMA

  const int tn = wave % tiles_n;    // 64-wide N tile
  const int tm = wave / tiles_n;    // 32-high M tile
  const int m0 = tm << 5;
  const int n0 = tn << 6;

  const int lane  = threadIdx.x & 31;
  const int lrow  = lane & 15;      // A: M row / B: N col
  const int lhalf = lane >> 4;      // K-half selector

  v8f acc[2][4] = {};

  const float* arow0 = A + (size_t)(m0 + lrow) * lda + (lhalf << 4);
  const float* arow1 = arow0 + (size_t)16 * lda;
  const float* wbase = W + (size_t)(n0 + lrow) * ldw + (lhalf << 4);

  for (int kb = 0; kb < K; kb += 32) {
    __builtin_prefetch(arow0 + kb + 32, 0, 0);
    __builtin_prefetch(wbase + kb + 32, 0, 0);

    v16h a0, a1;
#pragma unroll
    for (int j = 0; j < 8; j++) {
      a0[j]     = (_Float16)arow0[kb + j];
      a0[8 + j] = (_Float16)arow0[kb + 8 + j];
      a1[j]     = (_Float16)arow1[kb + j];
      a1[8 + j] = (_Float16)arow1[kb + 8 + j];
    }

#pragma unroll
    for (int s = 0; s < 4; s++) {
      const float* bp = wbase + (size_t)(s * 16) * ldw + kb;
      v16h bf;
#pragma unroll
      for (int j = 0; j < 16; j++) bf[j] = (_Float16)bp[j];
      acc[0][s] = __builtin_amdgcn_wmma_f32_16x16x32_f16(
          false, a0, false, bf, (short)0, acc[0][s], false, false);
      acc[1][s] = __builtin_amdgcn_wmma_f32_16x16x32_f16(
          false, a1, false, bf, (short)0, acc[1][s], false, false);
    }
  }

#pragma unroll
  for (int mi = 0; mi < 2; mi++) {
#pragma unroll
    for (int s = 0; s < 4; s++) {
#pragma unroll
      for (int r = 0; r < 8; r++) {
        C[(size_t)(m0 + 16 * mi + r + 8 * lhalf) * ldc + n0 + s * 16 + lrow] =
            acc[mi][s][r];
      }
    }
  }
}

// ============================================================
// Depthwise causal conv (k=4) + bias + SiLU.
// ============================================================
__global__ void conv_silu_kernel(const float* __restrict__ xz,
                                 const float* __restrict__ cw,
                                 const float* __restrict__ cb,
                                 float* __restrict__ xc) {
  const int idx = blockIdx.x * 256 + threadIdx.x;
  if (idx >= NROWS * D_INNER) return;
  const int d   = idx & (D_INNER - 1);
  const int row = idx >> 11;          // b*L + l   (D_INNER = 2048 = 2^11)
  const int l   = row & (LSEQ - 1);
  const int b   = row >> 11;          // LSEQ = 2048 = 2^11
  float acc = cb[d];
#pragma unroll
  for (int k = 0; k < D_CONV; k++) {
    const int ll = l - (D_CONV - 1) + k;
    if (ll >= 0)
      acc += xz[(size_t)(b * LSEQ + ll) * (2 * D_INNER) + d] * cw[d * D_CONV + k];
  }
  acc = acc * sigmoid_f(acc);
  xc[(size_t)row * D_INNER + d] = acc;
}

// ============================================================
// Chunked selective scan.
// Recurrence per (d, n):  h <- dA*h + du*B_n  is affine, so:
//   phase1: per (b, d, chunk) compose the 64-step affine map (P, Q) per n
//   phase2: per (b, d, n) combine the 32 chunk maps -> chunk-entry states H0
//   phase3: replay each chunk from H0, emitting gated y (fused softplus,
//           +u*D, *silu(z)), written in place over xc.
// Block = 256 d-channels of one (b, chunk); B/C staged in LDS per step.
// blockIdx.x = ((b*NCHUNK)+c)*8 + dgroup   (grid = 4*32*8 = 1024)
// ============================================================
__global__ void scan_phase1(const float* __restrict__ xc,
                            const float* __restrict__ dtp,
                            const float* __restrict__ dt_b,
                            const float* __restrict__ A_log,
                            const float* __restrict__ xdbl,
                            float* __restrict__ P,
                            float* __restrict__ Q) {
  const int bid = blockIdx.x;
  const int dg  = bid & 7;
  const int c   = (bid >> 3) & (NCHUNK - 1);
  const int b   = bid >> 8;
  const int d   = (dg << 8) + threadIdx.x;

  float p[D_STATE], q[D_STATE], Arow[D_STATE];
#pragma unroll
  for (int n = 0; n < D_STATE; n++) {
    p[n] = 1.f;
    q[n] = 0.f;
    Arow[n] = -__expf(A_log[d * D_STATE + n]);
  }
  const float bias = dt_b[d];

  __shared__ float sB[D_STATE];

  for (int l = c * CLEN; l < (c + 1) * CLEN; l++) {
    const int row = b * LSEQ + l;
    if (threadIdx.x < 16)
      sB[threadIdx.x] = xdbl[(size_t)row * XPROJ_LD + DT_RANK + threadIdx.x];
    __syncthreads();

    const size_t off = (size_t)row * D_INNER + d;
    const float u   = xc[off];
    const float dtv = softplus_f(dtp[off] + bias);
    const float du  = dtv * u;
#pragma unroll
    for (int n = 0; n < D_STATE; n++) {
      const float dA = __expf(dtv * Arow[n]);
      p[n] *= dA;
      q[n] = dA * q[n] + du * sB[n];
    }
    __syncthreads();
  }

  const size_t base = ((size_t)(b * NCHUNK + c) * D_INNER + d) * D_STATE;
#pragma unroll
  for (int n = 0; n < D_STATE; n++) {
    P[base + n] = p[n];
    Q[base + n] = q[n];
  }
}

__global__ void scan_phase2(const float* __restrict__ P,
                            const float* __restrict__ Q,
                            float* __restrict__ H0) {
  const int idx = blockIdx.x * 256 + threadIdx.x;   // over 4*2048*16 = 131072
  const int n = idx & (D_STATE - 1);
  const int d = (idx >> 4) & (D_INNER - 1);
  const int b = idx >> 15;
  float h = 0.f;
  for (int c = 0; c < NCHUNK; c++) {
    const size_t base = ((size_t)(b * NCHUNK + c) * D_INNER + d) * D_STATE + n;
    H0[base] = h;                 // state entering chunk c
    h = P[base] * h + Q[base];
  }
}

__global__ void scan_phase3(float* __restrict__ xc,          // in: conv(x); out: gated y
                            const float* __restrict__ dtp,
                            const float* __restrict__ dt_b,
                            const float* __restrict__ A_log,
                            const float* __restrict__ Dvec,
                            const float* __restrict__ xz,    // z branch
                            const float* __restrict__ xdbl,  // B/C (ld=192)
                            const float* __restrict__ H0) {
  const int bid = blockIdx.x;
  const int dg  = bid & 7;
  const int c   = (bid >> 3) & (NCHUNK - 1);
  const int b   = bid >> 8;
  const int d   = (dg << 8) + threadIdx.x;

  float h[D_STATE], Arow[D_STATE];
  const size_t base = ((size_t)(b * NCHUNK + c) * D_INNER + d) * D_STATE;
#pragma unroll
  for (int n = 0; n < D_STATE; n++) {
    h[n] = H0[base + n];
    Arow[n] = -__expf(A_log[d * D_STATE + n]);
  }
  const float Dd   = Dvec[d];
  const float bias = dt_b[d];

  __shared__ float sB[D_STATE];
  __shared__ float sC[D_STATE];

  for (int l = c * CLEN; l < (c + 1) * CLEN; l++) {
    const int row = b * LSEQ + l;
    if (threadIdx.x < 16)
      sB[threadIdx.x] = xdbl[(size_t)row * XPROJ_LD + DT_RANK + threadIdx.x];
    else if (threadIdx.x < 32)
      sC[threadIdx.x - 16] = xdbl[(size_t)row * XPROJ_LD + DT_RANK + D_STATE + (threadIdx.x - 16)];
    __syncthreads();

    const size_t off = (size_t)row * D_INNER + d;
    const float u   = xc[off];
    const float dtv = softplus_f(dtp[off] + bias);
    const float du  = dtv * u;
    float y = 0.f;
#pragma unroll
    for (int n = 0; n < D_STATE; n++) {
      h[n] = __expf(dtv * Arow[n]) * h[n] + du * sB[n];
      y += h[n] * sC[n];
    }
    const float z = xz[(size_t)row * (2 * D_INNER) + D_INNER + d];
    xc[off] = (y + u * Dd) * (z * sigmoid_f(z));
    __syncthreads();
  }
}

// ============================================================
// host-side launch
// ============================================================
extern "C" void kernel_launch(void* const* d_in, const int* in_sizes, int n_in,
                              void* d_out, int out_size, void* d_ws, size_t ws_size,
                              hipStream_t stream) {
  (void)in_sizes; (void)n_in; (void)out_size; (void)ws_size;

  const float* x         = (const float*)d_in[0];
  const float* norm_w    = (const float*)d_in[1];
  const float* in_proj_w = (const float*)d_in[2];   // [4096, 1024]
  const float* conv_w    = (const float*)d_in[3];   // [2048, 4]
  const float* conv_b    = (const float*)d_in[4];
  const float* x_proj_w  = (const float*)d_in[5];   // [160, 2048]
  const float* dt_proj_w = (const float*)d_in[6];   // [2048, 128]
  const float* dt_proj_b = (const float*)d_in[7];
  const float* A_log     = (const float*)d_in[8];   // [2048, 16]
  const float* Dvec      = (const float*)d_in[9];
  const float* out_proj_w= (const float*)d_in[10];  // [1024, 2048]
  const float* norm_f_w  = (const float*)d_in[11];
  float* out = (float*)d_out;

  // workspace carve-up (floats)
  const size_t PQ_ELEMS = (size_t)NBATCH * NCHUNK * D_INNER * D_STATE; // 4,194,304
  float* ws    = (float*)d_ws;
  float* xn    = ws;                                   // 8192*1024 (dead after in_proj)
  float* xz    = xn    + (size_t)NROWS * D_MODEL;      // 8192*4096
  float* xconv = xz    + (size_t)NROWS * 2 * D_INNER;  // 8192*2048 (becomes gated y)
  float* xdbl  = xconv + (size_t)NROWS * D_INNER;      // 8192*192 (padded ld)
  float* dtp   = xdbl  + (size_t)NROWS * XPROJ_LD;     // 8192*2048
  float* wxpad = dtp   + (size_t)NROWS * D_INNER;      // 192*2048 padded x_proj_w
  float* H0    = wxpad + (size_t)XPROJ_LD * D_INNER;   // 4,194,304
  float* Pbuf  = xn;                                   // reuse dead xn: P + Q fit exactly
  float* Qbuf  = xn + PQ_ELEMS;
  float* c2    = xn;                                   // reuse again after scan for out_proj

  auto launch_gemm = [&](const float* A, const float* W, float* C,
                         int M, int N, int K, int lda, int ldw, int ldc) {
    const int tiles_n = N / 64;
    const int total   = (M / 32) * tiles_n;
    const int blocks  = (total + 7) / 8;  // 8 waves per 256-thread block
    gemm_wmma_kernel<<<blocks, 256, 0, stream>>>(A, W, C, K, lda, ldw, ldc,
                                                 total, tiles_n);
  };

  // 0. pad x_proj_w to 192 rows (zeros beyond 160)
  pad_w_kernel<<<(XPROJ_LD * D_INNER) / 256, 256, 0, stream>>>(x_proj_w, wxpad);

  // 1. pre-norm
  rmsnorm_kernel<<<NROWS, 256, 0, stream>>>(x, norm_w, xn);

  // 2. in_proj: xz = xn @ in_proj_w.T   [8192,4096]
  launch_gemm(xn, in_proj_w, xz, NROWS, 2 * D_INNER, D_MODEL,
              D_MODEL, D_MODEL, 2 * D_INNER);

  // 3. depthwise causal conv + SiLU -> xconv [8192,2048]
  conv_silu_kernel<<<(NROWS * D_INNER) / 256, 256, 0, stream>>>(xz, conv_w, conv_b, xconv);

  // 4. x_proj (padded): xdbl = xconv @ wxpad.T   [8192,192]
  launch_gemm(xconv, wxpad, xdbl, NROWS, XPROJ_LD, D_INNER,
              D_INNER, D_INNER, XPROJ_LD);

  // 5. dt_proj (pre-activation): dtp = xdbl[:, :128] @ dt_proj_w.T  [8192,2048]
  launch_gemm(xdbl, dt_proj_w, dtp, NROWS, D_INNER, DT_RANK,
              XPROJ_LD, DT_RANK, D_INNER);

  // 6. chunked selective scan (xconv -> gated y in place)
  scan_phase1<<<NBATCH * NCHUNK * 8, 256, 0, stream>>>(
      xconv, dtp, dt_proj_b, A_log, xdbl, Pbuf, Qbuf);
  scan_phase2<<<(NBATCH * D_INNER * D_STATE) / 256, 256, 0, stream>>>(
      Pbuf, Qbuf, H0);
  scan_phase3<<<NBATCH * NCHUNK * 8, 256, 0, stream>>>(
      xconv, dtp, dt_proj_b, A_log, Dvec, xz, xdbl, H0);

  // 7. out_proj: c2 = y_gated @ out_proj_w.T   [8192,1024]
  launch_gemm(xconv, out_proj_w, c2, NROWS, D_MODEL, D_INNER,
              D_INNER, D_INNER, D_MODEL);

  // 8. residual add + final RMSNorm
  add_rmsnorm_kernel<<<NROWS, 256, 0, stream>>>(c2, x, norm_f_w, out);
}
